// FiSHAttention_58110907515412
// MI455X (gfx1250) — matrix operand order repaired and checked by reading, not compiled, not run on hardware
//
#include <hip/hip_runtime.h>
#include <stdint.h>

typedef __attribute__((ext_vector_type(16))) __bf16    v16bf;
typedef __attribute__((ext_vector_type(8)))  __bf16    v8bf;
typedef __attribute__((ext_vector_type(8)))  float     v8f;
typedef __attribute__((ext_vector_type(4)))  uint32_t  u32x4;
typedef __attribute__((ext_vector_type(8)))  uint32_t  u32x8;

#define WMMA_BF16(A, B, C) \
  __builtin_amdgcn_wmma_f32_16x16x32_bf16(false, (A), false, (B), (short)0, (C), false, false)

// A-fragment (16x32 bf16, M=lane%16): elements 0..7 = K[kb..kb+7], 8..15 = K[16+kb..16+kb+7],
// kb = (lane/16)*8.  Two contiguous 16B loads from a row-major row.
__device__ __forceinline__ v16bf load_afrag(const __bf16* rowp_k0, int kb) {
  v8bf lo = *(const v8bf*)(rowp_k0 + kb);
  v8bf hi = *(const v8bf*)(rowp_k0 + kb + 16);
  return __builtin_shufflevector(lo, hi, 0,1,2,3,4,5,6,7,8,9,10,11,12,13,14,15);
}

// ---------------------------------------------------------------------------
// Generic C = A(MxK, bf16 row-major) * B(KxN, bf16 K-major) tiled WMMA GEMM.
// Block = 256 threads = 8 waves as 4x2 wave grid -> 64x32 output tile.
// MODE 0: f32 store + bias, MODE 1: bf16 store, MODE 2: bf16 transposed store.
// ---------------------------------------------------------------------------
template <int MODE>
__global__ void __launch_bounds__(256)
gemm_nt(const __bf16* __restrict__ A, int lda,
        const __bf16* __restrict__ Bw, int ldb,
        void* __restrict__ Cv, int ldc,
        const float* __restrict__ bias, int Kd) {
  const int tid = threadIdx.x, wv = tid >> 5, lane = tid & 31;
  const int wr = wv >> 1, wc = wv & 1;
  const int m = lane & 15, hl = lane >> 4, kb = hl * 8;
  const int row0 = blockIdx.y * 64 + wr * 16;
  const int col0 = blockIdx.x * 32 + wc * 16;

  v8f acc = {};
  const __bf16* ap = A + (size_t)(row0 + m) * lda;
  for (int k0 = 0; k0 < Kd; k0 += 32) {
    v16bf af = load_afrag(ap + k0, kb);
    // B fragment: lane holds N column (col0+m), elements over K in A's table order.
    v16bf bf_;
#pragma unroll
    for (int s = 0; s < 16; s++) bf_[s] = Bw[(size_t)(k0 + kb + (s & 7) + ((s >> 3) << 4)) * ldb + col0 + m];
    acc = WMMA_BF16(af, bf_, acc);
  }

  if (MODE == 0) {
    float* C = (float*)Cv;
    float bb = bias ? bias[col0 + m] : 0.0f;
#pragma unroll
    for (int r = 0; r < 8; r++)
      C[(size_t)(row0 + r + 8 * hl) * ldc + col0 + m] = acc[r] + bb;
  } else if (MODE == 1) {
    __bf16* C = (__bf16*)Cv;
#pragma unroll
    for (int r = 0; r < 8; r++)
      C[(size_t)(row0 + r + 8 * hl) * ldc + col0 + m] = (__bf16)acc[r];
  } else {
    // Transposed store: C_T[col][row], 8 contiguous bf16 per lane = one 16B store.
    __bf16* C = (__bf16*)Cv;
    v8bf pk;
#pragma unroll
    for (int r = 0; r < 8; r++) pk[r] = (__bf16)acc[r];
    *(v8bf*)(C + (size_t)(col0 + m) * ldc + row0 + 8 * hl) = pk;
  }
}

// ---------------------------------------------------------------------------
// TDM: DMA one K-tile (256 dims x 64 tokens, bf16) from Kt[256][8192] into LDS.
// D# per CDNA5 ISA: group0 = {count, lds_addr, global_addr,
// type=2}; group1 = {data_size=2B, tensor_dim0=8192 (tokens, stride 8192),
// tensor_dim1=256 (dims), tile_dim0=64, tile_dim1=256}. 2D tensor -> 2-group
// form (VADDR2/VADDR3 = NULL). TDM writes LDS row-major [dim][64 tok], which
// is exactly the K-major WMMA B-fragment layout. Tracked by TENSORcnt.
// ---------------------------------------------------------------------------
__device__ __forceinline__ void tdm_load_kt_tile(const __bf16* Kt, int b, int jt,
                                                 uint32_t lds_byte_addr) {
  uint64_t ga = (uint64_t)(uintptr_t)Kt + 2ull * ((uint64_t)b * 1024u + (uint64_t)jt * 64u);
  u32x4 g0 = { 1u,                                   // count=1, user descriptor
               lds_byte_addr,                        // lds_addr (bytes)
               (uint32_t)ga,                         // global_addr[31:0]
               (uint32_t)((ga >> 32) & 0x01FFFFFFull) | 0x80000000u };  // addr[56:32] | type=2
  u32x8 g1 = { 1u << 16,                             // data_size=1 -> 2 bytes
               (8192u & 0xFFFFu) << 16,              // tensor_dim0[15:0] in [31:16]
               ((256u & 0xFFFFu) << 16) | (8192u >> 16),  // tensor_dim1 lo | tensor_dim0 hi
               (64u << 16) | (256u >> 16),           // tile_dim0=64 | tensor_dim1 hi
               256u,                                 // tile_dim2=0 | tile_dim1=256
               8192u,                                // tensor_dim0_stride[31:0]
               0u, 0u };                             // dim1_stride (unused for 2D)
  asm volatile("tensor_load_to_lds %0, %1" :: "s"(g0), "s"(g1) : "memory");
}

// ---------------------------------------------------------------------------
// Fused attention: one block = (batch b, head h, 32-row tile).
// S(32x1024) = Qh(32x256) @ Kcat^T via WMMA (mix[h,k]/8 folded into Q frags),
// K tiles staged by the Tensor Data Mover (double-buffered, TENSORcnt),
// exact softmax in LDS, O(32x64) = P @ V_h via WMMA. 8 waves as 2x4 grid.
// Inner loops are software-pipelined one fragment ahead so DS/VMEM latency
// hides under the previous WMMA instead of s_wait_dscnt-0 serialization.
// LDS: 128K scores + 65K probs + 2x32K K-staging ~= 257KB of the 320KB/WGP.
// ---------------------------------------------------------------------------
__global__ void __launch_bounds__(256)
attn_kernel(const __bf16* __restrict__ Q,    // [8192][256] row-major
            const __bf16* __restrict__ Kt,   // [256][8192] dim-major (pre-transposed)
            const __bf16* __restrict__ V,    // [8192][768] row-major
            const float*  __restrict__ mixs, // [12][4], softmaxed * 0.125
            __bf16* __restrict__ ctx) {      // [8192][768]
  __shared__ float  Sl[32][1024];
  __shared__ __bf16 Pl[32][1040];
  __shared__ __bf16 Kst[2][256 * 64];

  const int tid = threadIdx.x, wv = tid >> 5, lane = tid & 31;
  const int wr = wv >> 2, wc = wv & 3;           // 2x4 wave grid over 32x64 tiles
  const int m = lane & 15, hl = lane >> 4, kb = hl * 8;
  const int it = blockIdx.x & 31;
  const int h  = (blockIdx.x >> 5) % 12;
  const int b  = blockIdx.x / (32 * 12);
  const int rowg = b * 1024 + it * 32;

  // Generic LDS pointers: low 32 bits are the workgroup-relative LDS offset.
  const uint32_t kst_base0 = (uint32_t)(uintptr_t)&Kst[0][0];
  const uint32_t kst_base1 = (uint32_t)(uintptr_t)&Kst[1][0];

  // Kick off the first K-tile DMA immediately: it overlaps the Q preload below.
  if (wv == 0) tdm_load_kt_tile(Kt, b, 0, kst_base0);

  // Preload this wave's Q row-band (16x256) as 8 scaled A-fragments (64 VGPRs).
  v16bf qf[8];
  const __bf16* qrow = Q + (size_t)(rowg + wr * 16 + m) * 256;
#pragma unroll
  for (int ks = 0; ks < 8; ks++) {
    v16bf a = load_afrag(qrow + ks * 32, kb);
    float s = mixs[h * 4 + (ks >> 1)];  // k-group = (ks*32)/64
#pragma unroll
    for (int e = 0; e < 16; e++) a[e] = (__bf16)((float)a[e] * s);
    qf[ks] = a;
  }

  // S = Qh @ Kcat^T with TDM-staged K tiles (issue-ahead by one tile).
  for (int jt = 0; jt < 16; jt++) {
    if (wv == 0) __builtin_amdgcn_s_wait_tensorcnt(0);  // tile jt landed in LDS
    __syncthreads();                                    // publish; prior reads of next buf done
    if (wv == 0 && jt < 15)
      tdm_load_kt_tile(Kt, b, jt + 1, (jt & 1) ? kst_base0 : kst_base1);

    const __bf16* ktile = Kst[jt & 1];
    const int col0 = jt * 64 + wc * 16;
    v8f acc = {};
    // Pipelined: B-frag for ks+1 is in flight while WMMA for ks executes.
    v16bf kf;
#pragma unroll
    for (int s = 0; s < 16; s++) kf[s] = ktile[(kb + (s & 7) + ((s >> 3) << 4)) * 64 + wc * 16 + m];
#pragma unroll
    for (int ks = 0; ks < 7; ks++) {
      v16bf kfn;
#pragma unroll
      for (int s = 0; s < 16; s++) kfn[s] = ktile[((ks + 1) * 32 + kb + (s & 7) + ((s >> 3) << 4)) * 64 + wc * 16 + m];
      acc = WMMA_BF16(qf[ks], kf, acc);
      kf = kfn;
    }
    acc = WMMA_BF16(qf[7], kf, acc);
#pragma unroll
    for (int r = 0; r < 8; r++) Sl[wr * 16 + r + 8 * hl][col0 + m] = acc[r];
  }
  __syncthreads();

  // Exact softmax: wave wv owns rows 4wv..4wv+3; 32-lane shfl reductions.
  for (int rr = 0; rr < 4; rr++) {
    const int mr = wv * 4 + rr;
    float mx = -3.4e38f;
    for (int c = lane; c < 1024; c += 32) mx = fmaxf(mx, Sl[mr][c]);
    for (int off = 16; off; off >>= 1) mx = fmaxf(mx, __shfl_xor(mx, off));
    float sm = 0.0f;
    for (int c = lane; c < 1024; c += 32) sm += __expf(Sl[mr][c] - mx);
    for (int off = 16; off; off >>= 1) sm += __shfl_xor(sm, off);
    const float inv = 1.0f / sm;
    for (int c = lane; c < 1024; c += 32)
      Pl[mr][c] = (__bf16)(__expf(Sl[mr][c] - mx) * inv);
  }
  __syncthreads();

  // O = P @ V_h : A-frags from LDS probs, B-frags contiguous from global V.
  // Software-pipelined one k-step ahead (rotating af/bf registers).
  const __bf16* prow = &Pl[wr * 16 + m][0];
  const __bf16* vbase = V + (size_t)(b * 1024) * 768 + h * 64 + wc * 16 + m;
  v8f oacc = {};
  v16bf af  = load_afrag(prow, kb);
  v16bf bf_;
#pragma unroll
  for (int s = 0; s < 16; s++) bf_[s] = vbase[(size_t)(kb + (s & 7) + ((s >> 3) << 4)) * 768];
  for (int k0 = 0; k0 < 992; k0 += 32) {
    v16bf afn = load_afrag(prow + k0 + 32, kb);
    v16bf bfn;
#pragma unroll
    for (int s = 0; s < 16; s++) bfn[s] = vbase[(size_t)(k0 + 32 + kb + (s & 7) + ((s >> 3) << 4)) * 768];
    oacc = WMMA_BF16(af, bf_, oacc);
    af = afn; bf_ = bfn;
  }
  oacc = WMMA_BF16(af, bf_, oacc);
#pragma unroll
  for (int r = 0; r < 8; r++)
    ctx[(size_t)(rowg + wr * 16 + r + 8 * hl) * 768 + h * 64 + wc * 16 + m] = (__bf16)oacc[r];
}

// ---------------------------------------------------------------------------
// Small helpers: f32->bf16, f32->bf16 transposed (weights), mix softmax.
// ---------------------------------------------------------------------------
__global__ void cvt_bf16(const float* __restrict__ in, __bf16* __restrict__ out, int n) {
  int i = blockIdx.x * blockDim.x + threadIdx.x;
  if (i < n) out[i] = (__bf16)in[i];
}

__global__ void cvt_bf16_T(const float* __restrict__ Win, __bf16* __restrict__ Wt,
                           int R /*out ch*/, int Cc /*in dim*/) {
  int i = blockIdx.x * blockDim.x + threadIdx.x;
  if (i < R * Cc) {
    int c = i % R, d = i / R;
    Wt[(size_t)d * R + c] = (__bf16)Win[(size_t)c * Cc + d];
  }
}

__global__ void mix_softmax_k(const float* __restrict__ ml, float* __restrict__ mixs) {
  int h = threadIdx.x;
  if (h < 12) {
    float v0 = ml[h * 4], v1 = ml[h * 4 + 1], v2 = ml[h * 4 + 2], v3 = ml[h * 4 + 3];
    float mx = fmaxf(fmaxf(v0, v1), fmaxf(v2, v3));
    float e0 = __expf(v0 - mx), e1 = __expf(v1 - mx), e2 = __expf(v2 - mx), e3 = __expf(v3 - mx);
    float inv = 0.125f / (e0 + e1 + e2 + e3);  // fold HEAD_DIM^-0.5
    mixs[h * 4] = e0 * inv; mixs[h * 4 + 1] = e1 * inv;
    mixs[h * 4 + 2] = e2 * inv; mixs[h * 4 + 3] = e3 * inv;
  }
}

// ---------------------------------------------------------------------------
extern "C" void kernel_launch(void* const* d_in, const int* in_sizes, int n_in,
                              void* d_out, int out_size, void* d_ws, size_t ws_size,
                              hipStream_t stream) {
  (void)in_sizes; (void)n_in; (void)out_size; (void)ws_size;
  const float* x      = (const float*)d_in[0];
  const float* W_qkv  = (const float*)d_in[1];
  const float* W_v    = (const float*)d_in[2];
  const float* W_proj = (const float*)d_in[3];
  const float* b_proj = (const float*)d_in[4];
  const float* mixl   = (const float*)d_in[5];

  const size_t T = 8192;  // B*N tokens
  char* wsp = (char*)d_ws;
  auto carve = [&](size_t bytes) -> char* {
    char* p = wsp; wsp += (bytes + 255) & ~(size_t)255; return p;
  };
  __bf16* xb   = (__bf16*)carve(T * 768 * 2);        // x in bf16
  __bf16* wqkT = (__bf16*)carve(768ULL * 512 * 2);   // W_qk^T  [768 in][512 out]
  __bf16* wvT  = (__bf16*)carve(768ULL * 768 * 2);   // W_v^T
  __bf16* wpT  = (__bf16*)carve(768ULL * 768 * 2);   // W_proj^T
  __bf16* Qb   = (__bf16*)carve(T * 256 * 2);        // Q  [8192][256]
  __bf16* KtG  = (__bf16*)carve(256ULL * T * 2);     // K^T [256][8192]
  __bf16* Vb   = (__bf16*)carve(T * 768 * 2);        // V  [8192][768]
  __bf16* ctx  = (__bf16*)carve(T * 768 * 2);        // attention output
  float*  mixs = (float*)carve(48 * 4);

  cvt_bf16<<<(int)((T * 768 + 255) / 256), 256, 0, stream>>>(x, xb, (int)(T * 768));
  cvt_bf16_T<<<(512 * 768 + 255) / 256, 256, 0, stream>>>(W_qkv, wqkT, 512, 768);
  cvt_bf16_T<<<(768 * 768 + 255) / 256, 256, 0, stream>>>(W_v,   wvT,  768, 768);
  cvt_bf16_T<<<(768 * 768 + 255) / 256, 256, 0, stream>>>(W_proj, wpT, 768, 768);
  mix_softmax_k<<<1, 32, 0, stream>>>(mixl, mixs);

  // Q projection (normal store) and K projection (transposed store, dim-major)
  gemm_nt<1><<<dim3(8, 128),  256, 0, stream>>>(xb, 768, wqkT,       512, Qb,  256,  nullptr, 768);
  gemm_nt<2><<<dim3(8, 128),  256, 0, stream>>>(xb, 768, wqkT + 256, 512, KtG, 8192, nullptr, 768);
  // V projection
  gemm_nt<1><<<dim3(24, 128), 256, 0, stream>>>(xb, 768, wvT, 768, Vb, 768, nullptr, 768);
  // Fused mixed-head attention: 8 batches * 12 heads * 32 row-tiles
  attn_kernel<<<3072, 256, 0, stream>>>(Qb, KtG, Vb, mixs, ctx);
  // Output projection with bias -> f32 d_out
  gemm_nt<0><<<dim3(24, 128), 256, 0, stream>>>(ctx, 768, wpT, 768, d_out, 768, b_proj, 768);
}
